// LinkPredictionModel_SAGE_37709812859562
// MI455X (gfx1250) — matrix operand order repaired and checked by reading, not compiled, not run on hardware
//
#include <hip/hip_runtime.h>
#include <hip/hip_bf16.h>

#define NN  100000     // nodes
#define EE  1600000    // edges
#define DD  128        // feature dim (in == hid == out)
#define WPAD 136       // LDS row stride in bf16 elems (breaks 256B bank periodicity)

typedef __attribute__((ext_vector_type(16))) __bf16 v16bf;
typedef __attribute__((ext_vector_type(8)))  float  v8f;

// ---------------------------------------------------------------- utilities

__global__ __launch_bounds__(256)
void zero_kernel(float4* __restrict__ p, int n4) {
    int i = blockIdx.x * 256 + threadIdx.x;
    if (i < n4) p[i] = make_float4(0.f, 0.f, 0.f, 0.f);
}

__global__ __launch_bounds__(256)
void degree_kernel(const int* __restrict__ dst, float* __restrict__ cnt, int e) {
    int i = blockIdx.x * 256 + threadIdx.x;
    if (i < e) atomicAdd(&cnt[dst[i]], 1.0f);
}

// One wave per edge: 32 lanes x float4 = 128 features. Gather x[src], atomic
// scatter-add into agg[dst]. Both arrays are L2-resident (192MB L2).
__global__ __launch_bounds__(256)
void scatter_kernel(const float* __restrict__ feat, const int* __restrict__ src,
                    const int* __restrict__ dst, float* __restrict__ agg, int e_total) {
    long long t = (long long)blockIdx.x * 256 + threadIdx.x;
    int e = (int)(t >> 5);
    if (e >= e_total) return;
    int c = ((int)t & 31) * 4;
    int s = src[e], d = dst[e];
    const float4 v = *(const float4*)(feat + (size_t)s * DD + c);
    float* o = agg + (size_t)d * DD + c;
    atomicAdd(o + 0, v.x);
    atomicAdd(o + 1, v.y);
    atomicAdd(o + 2, v.z);
    atomicAdd(o + 3, v.w);
}

// ---------------------------------------------------------------- WMMA GEMM

// A fragment (16x32 bf16), contiguous-K order (matches B): lane l -> row M=l%16,
//   elems 0..7  <-> k = kbase..kbase+7        (kbase = ks*32 + (l/16)*16)
//   elems 8..15 <-> k = kbase+8..kbase+15
__device__ inline v16bf make_a_frag(const float* __restrict__ row, int kbase, float scale) {
    v16bf a;
    const float4 f0 = *(const float4*)(row + kbase);
    const float4 f1 = *(const float4*)(row + kbase + 4);
    const float4 f2 = *(const float4*)(row + kbase + 8);
    const float4 f3 = *(const float4*)(row + kbase + 12);
    a[0]  = (__bf16)(f0.x * scale); a[1]  = (__bf16)(f0.y * scale);
    a[2]  = (__bf16)(f0.z * scale); a[3]  = (__bf16)(f0.w * scale);
    a[4]  = (__bf16)(f1.x * scale); a[5]  = (__bf16)(f1.y * scale);
    a[6]  = (__bf16)(f1.z * scale); a[7]  = (__bf16)(f1.w * scale);
    a[8]  = (__bf16)(f2.x * scale); a[9]  = (__bf16)(f2.y * scale);
    a[10] = (__bf16)(f2.z * scale); a[11] = (__bf16)(f2.w * scale);
    a[12] = (__bf16)(f3.x * scale); a[13] = (__bf16)(f3.y * scale);
    a[14] = (__bf16)(f3.z * scale); a[15] = (__bf16)(f3.w * scale);
    return a;
}

// B fragment (32x16 bf16), contiguous-K order (matches A): lane l -> col N=l%16, contiguous
// K = (l/16)*16 .. +15 within this K-step -> one 32B contiguous LDS read.
__device__ inline v16bf load_b_frag(const __bf16* __restrict__ p) {
    union { v16bf v; int4 q[2]; } u;
    u.q[0] = *(const int4*)(p);
    u.q[1] = *(const int4*)(p + 8);
    return u.v;
}

// out[m, :] = [relu]( (agg[m]/max(cnt[m],1)) @ Wl^T + feat[m] @ Wr^T + bias )
// Block = 256 threads = 8 waves; wave w owns rows [blk*128 + w*16, +16).
__global__ __launch_bounds__(256)
void sage_gemm_kernel(const float* __restrict__ feat, const float* __restrict__ agg,
                      const float* __restrict__ cnt,  const float* __restrict__ Wl,
                      const float* __restrict__ Wr,   const float* __restrict__ bias,
                      float* __restrict__ out, int n, int do_relu) {
    __shared__ __bf16 sWl[DD * WPAD];
    __shared__ __bf16 sWr[DD * WPAD];

    const int tid = threadIdx.x;
    // Stage both weight matrices into LDS as bf16 (once per workgroup).
    for (int i = tid * 4; i < DD * DD; i += 256 * 4) {
        const int r = i >> 7, k = i & (DD - 1);
        const float4 l4 = *(const float4*)(Wl + i);
        const float4 r4 = *(const float4*)(Wr + i);
        __bf16* pl = sWl + r * WPAD + k;
        __bf16* pr = sWr + r * WPAD + k;
        pl[0] = (__bf16)l4.x; pl[1] = (__bf16)l4.y; pl[2] = (__bf16)l4.z; pl[3] = (__bf16)l4.w;
        pr[0] = (__bf16)r4.x; pr[1] = (__bf16)r4.y; pr[2] = (__bf16)r4.z; pr[3] = (__bf16)r4.w;
    }
    __syncthreads();

    const int lane  = tid & 31;
    const int wave  = tid >> 5;
    const int half  = lane >> 4;   // K half selector
    const int l16   = lane & 15;   // A: row-in-tile, B/D: col-in-tile
    const int mbase = blockIdx.x * 128 + wave * 16;
    if (mbase >= n) return;

    const int arow = mbase + l16 < n ? mbase + l16 : n - 1;
    const float deg   = cnt[arow];
    const float scale = 1.0f / fmaxf(deg, 1.0f);
    const float* frow = feat + (size_t)arow * DD;
    const float* grow = agg  + (size_t)arow * DD;

    // Preload all K-steps of both A operands (mean & self).
    v16bf am[4], ax[4];
#pragma unroll
    for (int ks = 0; ks < 4; ++ks) {
        const int kb = ks * 32 + half * 16;
        am[ks] = make_a_frag(grow, kb, scale);
        ax[ks] = make_a_frag(frow, kb, 1.0f);
    }

#pragma unroll
    for (int nt = 0; nt < 8; ++nt) {
        const float bv = bias[nt * 16 + l16];
        v8f c = { bv, bv, bv, bv, bv, bv, bv, bv };
#pragma unroll
        for (int ks = 0; ks < 4; ++ks) {
            const int boff = (nt * 16 + l16) * WPAD + ks * 32 + half * 16;
            const v16bf bl = load_b_frag(sWl + boff);
            const v16bf br = load_b_frag(sWr + boff);
            c = __builtin_amdgcn_wmma_f32_16x16x32_bf16(false, am[ks], false, bl,
                                                        (short)0, c, false, false);
            c = __builtin_amdgcn_wmma_f32_16x16x32_bf16(false, ax[ks], false, br,
                                                        (short)0, c, false, false);
        }
        // D layout: VGPR r, lane l -> row = r + 8*(l/16), col = l%16
#pragma unroll
        for (int r = 0; r < 8; ++r) {
            const int row = mbase + r + half * 8;
            float v = c[r];
            if (do_relu) v = fmaxf(v, 0.0f);
            if (row < n) out[(size_t)row * DD + nt * 16 + l16] = v;
        }
    }
}

// ---------------------------------------------------------------- launcher

extern "C" void kernel_launch(void* const* d_in, const int* in_sizes, int n_in,
                              void* d_out, int out_size, void* d_ws, size_t ws_size,
                              hipStream_t stream) {
    const float* x   = (const float*)d_in[0];
    const int*   ei  = (const int*)d_in[1];
    const int*   src = ei;
    const int*   dst = ei + EE;
    const float* W1l = (const float*)d_in[2];
    const float* W1r = (const float*)d_in[3];
    const float* b1  = (const float*)d_in[4];
    const float* W2l = (const float*)d_in[5];
    const float* W2r = (const float*)d_in[6];
    const float* b2  = (const float*)d_in[7];
    float* out = (float*)d_out;

    float* agg = (float*)d_ws;                       // N*128 floats
    float* cnt = agg + (size_t)NN * DD;              // N floats
    float* h   = cnt + NN;                           // N*128 floats

    const int agg4 = (NN * DD) / 4;                  // float4 counts
    const int ac4  = (NN * DD + NN) / 4;             // agg + cnt contiguous

    const int zb_ac  = (ac4 + 255) / 256;
    const int zb_agg = (agg4 + 255) / 256;
    const int eb     = (EE + 255) / 256;
    const int sb     = (int)(((long long)EE * 32) / 256);
    const int gb     = (NN + 127) / 128;

    // Layer 1
    zero_kernel<<<zb_ac, 256, 0, stream>>>((float4*)agg, ac4);
    degree_kernel<<<eb, 256, 0, stream>>>(dst, cnt, EE);
    scatter_kernel<<<sb, 256, 0, stream>>>(x, src, dst, agg, EE);
    sage_gemm_kernel<<<gb, 256, 0, stream>>>(x, agg, cnt, W1l, W1r, b1, h, NN, 1);

    // Layer 2
    zero_kernel<<<zb_agg, 256, 0, stream>>>((float4*)agg, agg4);
    scatter_kernel<<<sb, 256, 0, stream>>>(h, src, dst, agg, EE);
    sage_gemm_kernel<<<gb, 256, 0, stream>>>(h, agg, cnt, W2l, W2r, b2, out, NN, 0);
}
